// LinearAttention_62783831933558
// MI455X (gfx1250) — matrix operand order repaired and checked by reading, not compiled, not run on hardware
//
#include <hip/hip_runtime.h>

// Problem dims (fixed by reference)
#define B_ 32
#define N_ 1024
#define C_ 768
#define H_ 12
#define E_ 64
static __device__ __constant__ float SCALE_ = 0.125f;  // 64^-0.5
static __device__ __constant__ float EPS_ = 1e-6f;

typedef __attribute__((ext_vector_type(16))) __bf16 v16bf;
typedef __attribute__((ext_vector_type(8)))  float  v8f;
typedef __attribute__((ext_vector_type(4)))  float  f32x4;
typedef __attribute__((ext_vector_type(2)))  unsigned int u32x2;
typedef __attribute__((ext_vector_type(4)))  unsigned int u32x4;

union FragBF { v16bf v; u32x4 q[2]; __bf16 h[16]; };
union Pack4  { __bf16 h[4]; u32x2 u; };
union Pack8  { __bf16 h[8]; u32x4 u; };
union BF16U  { __bf16 b; unsigned short s; };

// bf16x3: D += Ahi*Bhi + Alo*Bhi + Ahi*Blo  (lo*lo term ~2^-16, dropped)
__device__ __forceinline__ v8f wmma3(const v16bf& ah, const v16bf& al,
                                     const v16bf& bh, const v16bf& bl, v8f c) {
  c = __builtin_amdgcn_wmma_f32_16x16x32_bf16(false, ah, false, bh, (short)0, c, false, false);
  c = __builtin_amdgcn_wmma_f32_16x16x32_bf16(false, al, false, bh, (short)0, c, false, false);
  c = __builtin_amdgcn_wmma_f32_16x16x32_bf16(false, ah, false, bl, (short)0, c, false, false);
  return c;
}

__device__ __forceinline__ void split(float x, __bf16& hi, __bf16& lo) {
  hi = (__bf16)x;                 // RNE
  lo = (__bf16)(x - (float)hi);   // residual
}

__device__ __forceinline__ void nt_store_bf16(__bf16 v, __bf16* p) {
  BF16U u; u.b = v;
  __builtin_nontemporal_store(u.s, (unsigned short*)p);
}

// generic -> LDS(as3) 32-bit address for async-to-LDS VDST operand
typedef __attribute__((address_space(3))) const void lds_cv;
__device__ __forceinline__ uint32_t lds_addr(const void* p) {
  return (uint32_t)(uintptr_t)(lds_cv*)p;
}

// per-lane 16B async copy global -> LDS (ASYNCcnt-tracked, no VGPR data path)
__device__ __forceinline__ void async_b128(uint32_t lds_off, const void* g) {
  asm volatile("global_load_async_to_lds_b128 %0, %1, off"
               :: "v"(lds_off), "v"(g) : "memory");
}
__device__ __forceinline__ void wait_async_le8() {
  asm volatile("s_wait_asynccnt 0x8" ::: "memory");
}
__device__ __forceinline__ void wait_async_0() {
  asm volatile("s_wait_asynccnt 0x0" ::: "memory");
}

// ---------------------------------------------------------------------------
// Elementwise f32 -> (bf16 hi, bf16 lo) split, float4 granularity.
// NT load (single use); stores stay cached (consumer GEMM follows at once).
// ---------------------------------------------------------------------------
__global__ __launch_bounds__(256) void split_kernel(const float* __restrict__ in,
                                                    __bf16* __restrict__ hi,
                                                    __bf16* __restrict__ lo,
                                                    long n4) {
  const long i = (long)blockIdx.x * blockDim.x + threadIdx.x;
  if (i >= n4) return;
  const f32x4 a = __builtin_nontemporal_load((const f32x4*)in + i);
  Pack4 ph, pl;
  split(a[0], ph.h[0], pl.h[0]); split(a[1], ph.h[1], pl.h[1]);
  split(a[2], ph.h[2], pl.h[2]); split(a[3], ph.h[3], pl.h[3]);
  ((u32x2*)hi)[i] = ph.u;
  ((u32x2*)lo)[i] = pl.u;
}

// ---------------------------------------------------------------------------
// Big GEMM on pre-split bf16: Y[M x Nd] = A * Bw^T (row-major, K contig).
// Double-buffered LDS staging via GLOBAL_LOAD_ASYNC_TO_LDS_B128.
// Grid: blockIdx.x = N-tile (weights stay L2-resident; consecutive blocks
// share the A slab), blockIdx.y = M-tile.
// MODE 0: qkv epilogue -> q (B,H,N,E) hi/lo; k_,v TRANSPOSED (B,H,E,N) hi/lo
//         (packed b128 NT stores: 8 consecutive n per lane).
// MODE 1: +bias -> f32 out, NT stores.
// ---------------------------------------------------------------------------
template <int MODE>
__global__ __launch_bounds__(256) void gemm_bf16x3(
    const __bf16* __restrict__ Ahg, const __bf16* __restrict__ Alg,
    const __bf16* __restrict__ Bhg, const __bf16* __restrict__ Blg,
    const float* __restrict__ bias,
    __bf16* __restrict__ qh, __bf16* __restrict__ ql,
    __bf16* __restrict__ kTh, __bf16* __restrict__ kTl,
    __bf16* __restrict__ vTh, __bf16* __restrict__ vTl,
    float* __restrict__ Yout, int K) {
  constexpr int BM = 128, BN = 128, BK = 32;
  constexpr int LDSS = 40;  // padded row stride (elems): 80B rows, 16B aligned
  __shared__ __bf16 Ah[2][BM * LDSS], Al[2][BM * LDSS];
  __shared__ __bf16 Bh[2][BN * LDSS], Bl[2][BN * LDSS];

  const int tid  = threadIdx.x;
  const int lane = tid & 31;
  const int wave = tid >> 5;
  const int bn = blockIdx.x * BN;  // N-tile fastest-varying
  const int bm = blockIdx.y * BM;
  const int wm = (wave >> 2) * 64;
  const int wn = (wave & 3) * 32;
  const int ml = lane & 15;
  const int hi16 = lane >> 4;

  v8f acc[4][2] = {};

  // 512 16B chunks per array; each thread moves 2 chunks x 4 arrays = 8 asyncs
  auto issue = [&](int k0, int p) {
#pragma unroll
    for (int j = 0; j < 2; ++j) {
      const int ch = tid + j * 256;
      const int row = ch >> 2;
      const int cb = (ch & 3) * 16;
      async_b128(lds_addr(&Ah[p][row * LDSS]) + cb,
                 (const char*)(Ahg + (size_t)(bm + row) * K + k0) + cb);
      async_b128(lds_addr(&Al[p][row * LDSS]) + cb,
                 (const char*)(Alg + (size_t)(bm + row) * K + k0) + cb);
      async_b128(lds_addr(&Bh[p][row * LDSS]) + cb,
                 (const char*)(Bhg + (size_t)(bn + row) * K + k0) + cb);
      async_b128(lds_addr(&Bl[p][row * LDSS]) + cb,
                 (const char*)(Blg + (size_t)(bn + row) * K + k0) + cb);
    }
  };

  const int T = K / BK;
  issue(0, 0);
  for (int i = 0; i < T; ++i) {
    const int p = i & 1;
    if (i + 1 < T) {           // overlap next tile's copies with this compute
      issue((i + 1) * BK, p ^ 1);
      wait_async_le8();        // in-order completion: tile i is in LDS
    } else {
      wait_async_0();
    }
    __syncthreads();

    FragBF afh[4], afl[4];
#pragma unroll
    for (int mt = 0; mt < 4; ++mt) {
      const int base = (wm + mt * 16 + ml) * LDSS + (hi16 << 4);
      afh[mt].q[0] = *(const u32x4*)&Ah[p][base];
      afh[mt].q[1] = *(const u32x4*)&Ah[p][base + 8];
      afl[mt].q[0] = *(const u32x4*)&Al[p][base];
      afl[mt].q[1] = *(const u32x4*)&Al[p][base + 8];
    }
#pragma unroll
    for (int nt = 0; nt < 2; ++nt) {
      const int base = (wn + nt * 16 + ml) * LDSS + (hi16 << 4);
      FragBF bfh, bfl;
      bfh.q[0] = *(const u32x4*)&Bh[p][base];
      bfh.q[1] = *(const u32x4*)&Bh[p][base + 8];
      bfl.q[0] = *(const u32x4*)&Bl[p][base];
      bfl.q[1] = *(const u32x4*)&Bl[p][base + 8];
#pragma unroll
      for (int mt = 0; mt < 4; ++mt)
        acc[mt][nt] = wmma3(afh[mt].v, afl[mt].v, bfh.v, bfl.v, acc[mt][nt]);
    }
    __syncthreads();
  }

  // Epilogue. D layout: VGPR r -> (M = r + 8*hi16, N = lane%16).
#pragma unroll
  for (int mt = 0; mt < 4; ++mt) {
#pragma unroll
    for (int nt = 0; nt < 2; ++nt) {
      const int d = bn + wn + nt * 16 + ml;
      const int m0 = bm + wm + mt * 16 + (hi16 << 3);  // 8 consecutive m
      if (MODE == 0) {
        const int s = d / C_;  // uniform per 16-wide tile (768 % 16 == 0)
        const int rem = d - s * C_;
        const int h = rem >> 6, e = rem & 63;
        const int b = m0 >> 10, nn0 = m0 & 1023;       // same b for all 8 r
        if (s == 0) {
          // q_: (B,H,N,E) layout (attn wants e-contiguous rows)
#pragma unroll
          for (int r = 0; r < 8; ++r) {
            __bf16 ph, pl;
            split(fmaxf(acc[mt][nt][r], 0.f) + SCALE_, ph, pl);
            const size_t off = (((size_t)(b * H_ + h)) * N_ + nn0 + r) * E_ + e;
            nt_store_bf16(ph, &qh[off]);
            nt_store_bf16(pl, &ql[off]);
          }
        } else {
          // k_/v: TRANSPOSED (B,H,E,N); lane's 8 r are consecutive n -> b128
          Pack8 ph8, pl8;
#pragma unroll
          for (int r = 0; r < 8; ++r) {
            const float val = (s == 1) ? fmaxf(acc[mt][nt][r], 0.f) + SCALE_
                                       : acc[mt][nt][r];
            split(val, ph8.h[r], pl8.h[r]);
          }
          const size_t off = (((size_t)(b * H_ + h)) * E_ + e) * N_ + nn0;
          if (s == 1) {
            __builtin_nontemporal_store(ph8.u, (u32x4*)&kTh[off]);
            __builtin_nontemporal_store(pl8.u, (u32x4*)&kTl[off]);
          } else {
            __builtin_nontemporal_store(ph8.u, (u32x4*)&vTh[off]);
            __builtin_nontemporal_store(pl8.u, (u32x4*)&vTl[off]);
          }
        }
      } else {
        const float bv = bias[d];
#pragma unroll
        for (int r = 0; r < 8; ++r)
          __builtin_nontemporal_store(acc[mt][nt][r] + bv,
                                      &Yout[(size_t)(m0 + r) * C_ + d]);
      }
    }
  }
}

// ---------------------------------------------------------------------------
// ksum[bh][e] = sum_n (k_hi + k_lo); kT rows are n-contiguous.
// ---------------------------------------------------------------------------
__global__ __launch_bounds__(64) void ksum_kernel(const __bf16* __restrict__ kTh,
                                                  const __bf16* __restrict__ kTl,
                                                  float* __restrict__ ksum) {
  const int bh = blockIdx.x, e = threadIdx.x;
  const __bf16* ph = kTh + ((size_t)bh * E_ + e) * N_;
  const __bf16* pl = kTl + ((size_t)bh * E_ + e) * N_;
  float s = 0.f;
  for (int n = 0; n < N_; ++n) s += (float)ph[n] + (float)pl[n];
  ksum[bh * E_ + e] = s;
}

// ---------------------------------------------------------------------------
// kv^T[bh][d][e] = sum_n k_[n][e] * v[n][d], stored pre-split bf16.
// All fragments are pure b128 loads (K = n contiguous in kT/vT).
// ---------------------------------------------------------------------------
__global__ __launch_bounds__(256) void kv_kernel(
    const __bf16* __restrict__ kTh, const __bf16* __restrict__ kTl,
    const __bf16* __restrict__ vTh, const __bf16* __restrict__ vTl,
    __bf16* __restrict__ kvTh, __bf16* __restrict__ kvTl) {
  const int bh = blockIdx.x;
  const int lane = threadIdx.x & 31, wave = threadIdx.x >> 5;
  const int et  = (wave >> 1) * 16;
  const int dt0 = (wave & 1) * 32;
  const int ml = lane & 15;
  const int kbA = (lane >> 4) * 8;
  const int kbB = (lane >> 4) * 16;

  const size_t slab = (size_t)bh * E_ * N_;
  const __bf16* ka = kTh + slab + (size_t)(et + ml) * N_;
  const __bf16* kb = kTl + slab + (size_t)(et + ml) * N_;
  const __bf16* v0h = vTh + slab + (size_t)(dt0 + ml) * N_;
  const __bf16* v0l = vTl + slab + (size_t)(dt0 + ml) * N_;
  const __bf16* v1h = v0h + (size_t)16 * N_;
  const __bf16* v1l = v0l + (size_t)16 * N_;
  v8f acc0 = {}, acc1 = {};

  for (int n0 = 0; n0 < N_; n0 += 32) {
    FragBF ah, al, b0h, b0l, b1h, b1l;
    ah.q[0] = *(const u32x4*)&ka[n0 + kbB];
    ah.q[1] = *(const u32x4*)&ka[n0 + kbB + 8];
    al.q[0] = *(const u32x4*)&kb[n0 + kbB];
    al.q[1] = *(const u32x4*)&kb[n0 + kbB + 8];
    b0h.q[0] = *(const u32x4*)&v0h[n0 + kbB];
    b0h.q[1] = *(const u32x4*)&v0h[n0 + kbB + 8];
    b0l.q[0] = *(const u32x4*)&v0l[n0 + kbB];
    b0l.q[1] = *(const u32x4*)&v0l[n0 + kbB + 8];
    b1h.q[0] = *(const u32x4*)&v1h[n0 + kbB];
    b1h.q[1] = *(const u32x4*)&v1h[n0 + kbB + 8];
    b1l.q[0] = *(const u32x4*)&v1l[n0 + kbB];
    b1l.q[1] = *(const u32x4*)&v1l[n0 + kbB + 8];
    acc0 = wmma3(ah.v, al.v, b0h.v, b0l.v, acc0);
    acc1 = wmma3(ah.v, al.v, b1h.v, b1l.v, acc1);
  }

  // store kv^T pre-split: lane's 8 r are consecutive e at fixed d -> b128
  const int mrow = et + kbA;  // kbA doubles as D-row offset (0 or 8)
  Pack8 h0, l0, h1, l1;
#pragma unroll
  for (int r = 0; r < 8; ++r) {
    split(acc0[r], h0.h[r], l0.h[r]);
    split(acc1[r], h1.h[r], l1.h[r]);
  }
  const size_t o0 = ((size_t)bh * E_ + dt0 + ml) * E_ + mrow;
  const size_t o1 = ((size_t)bh * E_ + dt0 + 16 + ml) * E_ + mrow;
  *(u32x4*)&kvTh[o0] = h0.u;  *(u32x4*)&kvTl[o0] = l0.u;
  *(u32x4*)&kvTh[o1] = h1.u;  *(u32x4*)&kvTl[o1] = l1.u;
}

// ---------------------------------------------------------------------------
// att = (q_ kv) * 1/(q_·ksum + eps), written pre-split (hi/lo bf16, (B,N,C)).
// A-fragments: b128 loads from q_ (e-contiguous). B: b128 from kv^T hi/lo.
// ---------------------------------------------------------------------------
__global__ __launch_bounds__(256) void attn_out_kernel(
    const __bf16* __restrict__ qhg, const __bf16* __restrict__ qlg,
    const __bf16* __restrict__ kvTh, const __bf16* __restrict__ kvTl,
    const float* __restrict__ ksum,
    __bf16* __restrict__ atth, __bf16* __restrict__ attl) {
  const int bh = blockIdx.x;
  const int b = bh / H_, h = bh % H_;
  const int m0 = blockIdx.y * 128;
  __shared__ float zsh[128];
  __shared__ float kssh[E_];

  const size_t qoff = (size_t)bh * N_ * E_ + (size_t)m0 * E_;
  const __bf16* qph = qhg + qoff;
  const __bf16* qpl = qlg + qoff;
  const int tid = threadIdx.x;

  if (tid < E_) kssh[tid] = ksum[bh * E_ + tid];
  __syncthreads();
  if (tid < 128) {
    float dv = 0.f;
    for (int e = 0; e < E_; ++e)
      dv += ((float)qph[(size_t)tid * E_ + e] + (float)qpl[(size_t)tid * E_ + e]) * kssh[e];
    zsh[tid] = 1.0f / (dv + EPS_);
  }
  __syncthreads();

  const int lane = tid & 31, wave = tid >> 5;
  const int mt = wave * 16;
  const int ml = lane & 15;
  const int kbA = (lane >> 4) * 8;
  const int kbB = (lane >> 4) * 16;

  v8f acc[4] = {};
#pragma unroll
  for (int e0 = 0; e0 < E_; e0 += 32) {
    FragBF ah, al;
    const size_t ro = (size_t)(mt + ml) * E_ + e0;
    ah.q[0] = *(const u32x4*)&qph[ro + kbB];
    ah.q[1] = *(const u32x4*)&qph[ro + kbB + 8];
    al.q[0] = *(const u32x4*)&qpl[ro + kbB];
    al.q[1] = *(const u32x4*)&qpl[ro + kbB + 8];
#pragma unroll
    for (int nt = 0; nt < 4; ++nt) {
      FragBF bfh, bfl;
      const size_t kb = ((size_t)bh * E_ + nt * 16 + ml) * E_ + e0 + kbB;
      bfh.q[0] = *(const u32x4*)&kvTh[kb];
      bfh.q[1] = *(const u32x4*)&kvTh[kb + 8];
      bfl.q[0] = *(const u32x4*)&kvTl[kb];
      bfl.q[1] = *(const u32x4*)&kvTl[kb + 8];
      acc[nt] = wmma3(ah.v, al.v, bfh.v, bfl.v, acc[nt]);
    }
  }

  const size_t abase = ((size_t)b * N_ + m0) * C_ + h * E_;
  const int mrow = mt + kbA;
#pragma unroll
  for (int nt = 0; nt < 4; ++nt)
#pragma unroll
    for (int r = 0; r < 8; ++r) {
      const int m = mrow + r;
      __bf16 ph, pl; split(acc[nt][r] * zsh[m], ph, pl);
      const size_t idx = abase + (size_t)m * C_ + nt * 16 + ml;
      atth[idx] = ph;   // cached: re-read as A by the projection GEMM
      attl[idx] = pl;
    }
}

// ---------------------------------------------------------------------------
extern "C" void kernel_launch(void* const* d_in, const int* in_sizes, int n_in,
                              void* d_out, int out_size, void* d_ws, size_t ws_size,
                              hipStream_t stream) {
  const float* x      = (const float*)d_in[0];
  const float* qkv_w  = (const float*)d_in[1];
  const float* proj_w = (const float*)d_in[2];
  const float* proj_b = (const float*)d_in[3];
  float* out = (float*)d_out;
  (void)in_sizes; (void)n_in; (void)out_size; (void)ws_size;

  const size_t per   = (size_t)B_ * N_ * C_;       // 25,165,824 (== B*H*N*E)
  const size_t wqkvN = (size_t)3 * C_ * C_;
  const size_t wprjN = (size_t)C_ * C_;
  const size_t kvN   = (size_t)B_ * H_ * E_ * E_;

  __bf16* p = (__bf16*)d_ws;
  __bf16* xh = p;   p += per;   // aliased as att hi (x split consumed first)
  __bf16* xl = p;   p += per;   // aliased as att lo
  __bf16* wqh = p;  p += wqkvN;
  __bf16* wql = p;  p += wqkvN;
  __bf16* wph = p;  p += wprjN;
  __bf16* wpl = p;  p += wprjN;
  __bf16* qh = p;   p += per;
  __bf16* ql = p;   p += per;
  __bf16* kTh = p;  p += per;   // TRANSPOSED (B,H,E,N)
  __bf16* kTl = p;  p += per;
  __bf16* vTh = p;  p += per;   // TRANSPOSED (B,H,E,N)
  __bf16* vTl = p;  p += per;
  __bf16* kvTh = p; p += kvN;   // kv^T (B,H,E(d),E(e)) pre-split
  __bf16* kvTl = p; p += kvN;
  float* ksum = (float*)p;

  dim3 blk(256);
  // 0) one-time bf16x2 splits
  split_kernel<<<dim3((unsigned)(per / 4 / 256)), blk, 0, stream>>>(x, xh, xl, per / 4);
  split_kernel<<<dim3((unsigned)(wqkvN / 4 / 256)), blk, 0, stream>>>(qkv_w, wqh, wql, wqkvN / 4);
  split_kernel<<<dim3((unsigned)(wprjN / 4 / 256)), blk, 0, stream>>>(proj_w, wph, wpl, wprjN / 4);
  // 1) QKV GEMM: M=32768, Nd=2304, K=768 (grid: N-tiles fastest)
  gemm_bf16x3<0><<<dim3((3 * C_) / 128, (B_ * N_) / 128), blk, 0, stream>>>(
      xh, xl, wqh, wql, nullptr, qh, ql, kTh, kTl, vTh, vTl, nullptr, C_);
  // 2) k_ row sums (contiguous in kT)
  ksum_kernel<<<dim3(B_ * H_), dim3(64), 0, stream>>>(kTh, kTl, ksum);
  // 3) kv^T = (k_^T v)^T per (b,h), pre-split
  kv_kernel<<<dim3(B_ * H_), blk, 0, stream>>>(kTh, kTl, vTh, vTl, kvTh, kvTl);
  // 4) att = (q_ kv) * z -> split att (aliases xh/xl, already consumed)
  attn_out_kernel<<<dim3(B_ * H_, N_ / 128), blk, 0, stream>>>(
      qh, ql, kvTh, kvTl, ksum, xh, xl);
  // 5) projection: M=32768, Nd=768, K=768 (+bias)
  gemm_bf16x3<1><<<dim3(C_ / 128, (B_ * N_) / 128), blk, 0, stream>>>(
      xh, xl, wph, wpl, proj_b, nullptr, nullptr, nullptr, nullptr, nullptr, nullptr,
      out, C_);
}
